// GroupQuantLinear_90933047591464
// MI455X (gfx1250) — matrix operand order repaired and checked by reading, not compiled, not run on hardware
//
#include <hip/hip_runtime.h>

// ---------------------------------------------------------------------------
// int4 group-quantized linear for MI455X (gfx1250, wave32, WMMA)
//   out[M, O] = x[M, K] * dequant(w)[O, K]^T
//   M = 512 (B*S), K = 8192, O = 12288, group size 64 along K.
//
// v3: software-pipelined, double-buffered LDS.
//   - ping-pong sA/sB, ONE barrier per K-group
//   - next tile's global loads issued before the WMMA block (latency hidden)
//   - x staged via v_cvt_pk_rtz_f16_f32; int4 dequant via fp16 bit trick
//     (0x6400|nib == 1024+nib) folded with v_pk_fma_f16
//   - 8 waves x (2x4) v_wmma_f32_16x16x32_f16, f32 accumulate
// ---------------------------------------------------------------------------

#define OUT_FEATURES 12288
#define IN_FEATURES  8192
#define GROUP_SIZE   64
#define N_GROUPS     (IN_FEATURES / GROUP_SIZE)   // 128
#define WORDS_PER_G  16                           // packed u16 words per group

#define BM 128
#define BN 128
#define BK 64                  // one quant group per K-step
#define LDS_STRIDE (BK + 8)    // 72 halves = 144 B row stride (16B aligned, bank skew)

typedef _Float16 v16h   __attribute__((ext_vector_type(16)));
typedef _Float16 h2     __attribute__((ext_vector_type(2)));
typedef __fp16   fp16x2 __attribute__((ext_vector_type(2)));   // cvt_pkrtz return type
typedef float    v8f    __attribute__((ext_vector_type(8)));

union FragH {
    uint4 q[2];   // 32 bytes
    v16h  v;
};
union H2U {
    unsigned u;
    h2       h;
    fp16x2   p;
};

__global__ __launch_bounds__(256)
void gq_linear_wmma_f16(const float* __restrict__ x,
                        const int*   __restrict__ w_packed,
                        const float* __restrict__ w_scale,
                        const float* __restrict__ w_bias,
                        float*       __restrict__ out,
                        int M) {
    __shared__ __align__(16) unsigned short sA[2][BM][LDS_STRIDE];  // x tiles (f16)
    __shared__ __align__(16) unsigned short sB[2][BN][LDS_STRIDE];  // w tiles (f16)

    const int t    = threadIdx.x;          // 0..255
    const int lane = t & 31;
    const int wid  = t >> 5;               // 0..7
    const int wm   = wid & 3;              // 4 waves along M  (4*32 = 128)
    const int wn   = wid >> 2;             // 2 waves along N  (2*64 = 128)

    const int n0 = blockIdx.x * BN;        // output-feature tile origin
    const int m0 = blockIdx.y * BM;        // row tile origin

    const int lhalf = lane >> 4;           // 0: lanes 0-15, 1: lanes 16-31
    const int l15   = lane & 15;

    // Per-thread staging coordinates (constant across K-groups).
    const int xrow  = t >> 4;              // wait: see loops below (per-i rows)
    (void)xrow;

    v8f acc[2][4];
    const v8f vzero = {0.f,0.f,0.f,0.f,0.f,0.f,0.f,0.f};
#pragma unroll
    for (int s = 0; s < 2; ++s)
#pragma unroll
        for (int n = 0; n < 4; ++n) acc[s][n] = vzero;

    // Staged registers for the in-flight (next) tile.
    float4 xr[8];
    int4   wpr[2];
    float  scr[2], bir[2];

    // ---- helper: issue global loads for K-group kb into registers ----------
    auto load_globals = [&](int kb) {
        const int k0 = kb * BK;
#pragma unroll
        for (int i = 0; i < 8; ++i) {
            const int f    = t + 256 * i;      // float4 id in 128x64 tile
            const int row  = f >> 4;           // 16 float4 per row
            const int col4 = f & 15;
            xr[i] = *(const float4*)&x[(size_t)(m0 + row) * IN_FEATURES + k0 + col4 * 4];
        }
#pragma unroll
        for (int s = 0; s < 2; ++s) {
            const int p   = t + 256 * s;       // block-of-16 id, 0..511
            const int row = p >> 2;            // 4 blocks per row
            const int b   = p & 3;
            const size_t wbase = (size_t)(n0 + row) * (N_GROUPS * WORDS_PER_G)
                               + (size_t)kb * WORDS_PER_G + b * 4;
            wpr[s] = *(const int4*)&w_packed[wbase];
            scr[s] = w_scale[(size_t)(n0 + row) * N_GROUPS + kb];
            bir[s] = w_bias [(size_t)(n0 + row) * N_GROUPS + kb];
        }
    };

    // ---- helper: convert/dequant staged registers into LDS buffer buf ------
    auto stage_lds = [&](int buf) {
#pragma unroll
        for (int i = 0; i < 8; ++i) {
            const int f    = t + 256 * i;
            const int row  = f >> 4;
            const int col4 = f & 15;
            H2U lo, hi;
            lo.p = __builtin_amdgcn_cvt_pkrtz(xr[i].x, xr[i].y);   // v_cvt_pk_rtz_f16_f32
            hi.p = __builtin_amdgcn_cvt_pkrtz(xr[i].z, xr[i].w);
            *(uint2*)&sA[buf][row][col4 * 4] = make_uint2(lo.u, hi.u);
        }
#pragma unroll
        for (int s = 0; s < 2; ++s) {
            const int p   = t + 256 * s;
            const int row = p >> 2;
            const int b   = p & 3;

            const _Float16 sh = (_Float16)scr[s];
            const _Float16 bh = (_Float16)(bir[s] - 1024.0f * scr[s]);
            const h2 s2 = {sh, sh};
            const h2 b2 = {bh, bh};

            const unsigned w0 = (unsigned)wpr[s].x, w1 = (unsigned)wpr[s].y;
            const unsigned w2 = (unsigned)wpr[s].z, w3 = (unsigned)wpr[s].w;

            unsigned tmpw[8];
#pragma unroll
            for (int i = 0; i < 4; ++i) {
                const unsigned wi = (i == 0) ? w0 : (i == 1) ? w1 : (i == 2) ? w2 : w3;
                const unsigned a0 = wi & 0xFu;
                const unsigned a1 = (wi >> 4) & 0xFu;
                const unsigned a2 = (wi >> 8) & 0xFu;
                const unsigned a3 = (wi >> 12) & 0xFu;
                H2U p01, p23, r01, r23;
                p01.u = 0x64006400u | a0 | (a1 << 16);   // fp16x2: 1024+nib
                p23.u = 0x64006400u | a2 | (a3 << 16);
                r01.h = __builtin_elementwise_fma(p01.h, s2, b2);  // v_pk_fma_f16
                r23.h = __builtin_elementwise_fma(p23.h, s2, b2);
                tmpw[2 * i]     = r01.u;
                tmpw[2 * i + 1] = r23.u;
            }
            *(uint4*)&sB[buf][row][b * 16 + 0] = *(uint4*)&tmpw[0];
            *(uint4*)&sB[buf][row][b * 16 + 8] = *(uint4*)&tmpw[4];
        }
    };

    // ---- helper: 16 WMMAs from LDS buffer buf ------------------------------
    auto compute = [&](int buf) {
#pragma unroll
        for (int ks = 0; ks < 2; ++ks) {
            const int kbase = ks * 32 + (lhalf ? 8 : 0);  // ISA 16-bit frag K interleave

            FragH aF[2];
#pragma unroll
            for (int s = 0; s < 2; ++s) {
                const int row = wm * 32 + s * 16 + l15;
                aF[s].q[0] = *(const uint4*)&sA[buf][row][kbase];
                aF[s].q[1] = *(const uint4*)&sA[buf][row][kbase + 16];
            }
            FragH bF[4];
#pragma unroll
            for (int n = 0; n < 4; ++n) {
                const int col = wn * 64 + n * 16 + l15;
                bF[n].q[0] = *(const uint4*)&sB[buf][col][kbase];
                bF[n].q[1] = *(const uint4*)&sB[buf][col][kbase + 16];
            }
#pragma unroll
            for (int s = 0; s < 2; ++s)
#pragma unroll
                for (int n = 0; n < 4; ++n)
                    acc[s][n] = __builtin_amdgcn_wmma_f32_16x16x32_f16(
                        false, aF[s].v, false, bF[n].v,
                        (short)0, acc[s][n], false, false);
        }
    };

    // ---- pipelined main loop: one barrier per K-group ----------------------
    load_globals(0);
    stage_lds(0);
    __syncthreads();

    for (int kb = 0; kb < N_GROUPS; ++kb) {
        const int cur = kb & 1;
        const int nxt = cur ^ 1;
        const bool has_next = (kb + 1) < N_GROUPS;

        if (has_next) load_globals(kb + 1);   // global_load_b128 clause issues here
        compute(cur);                          // WMMAs hide the load latency
        if (has_next) stage_lds(nxt);          // cvt/dequant into the other buffer
        __syncthreads();                       // single barrier: publishes nxt,
                                               // releases cur for the next write
    }

    // -------- epilogue: C layout = VGPR r -> M=r (lanes 0-15) / M=r+8 -------
#pragma unroll
    for (int s = 0; s < 2; ++s) {
#pragma unroll
        for (int n = 0; n < 4; ++n) {
            const int col = n0 + wn * 64 + n * 16 + l15;
#pragma unroll
            for (int r = 0; r < 8; ++r) {
                const int row = m0 + wm * 32 + s * 16 + r + lhalf * 8;
                out[(size_t)row * OUT_FEATURES + col] = acc[s][n][r];
            }
        }
    }
}

extern "C" void kernel_launch(void* const* d_in, const int* in_sizes, int n_in,
                              void* d_out, int out_size, void* d_ws, size_t ws_size,
                              hipStream_t stream) {
    const float* x        = (const float*)d_in[0];
    const int*   w_packed = (const int*)  d_in[1];
    const float* w_scale  = (const float*)d_in[2];
    const float* w_bias   = (const float*)d_in[3];
    float*       out      = (float*)d_out;

    const int M = in_sizes[0] / IN_FEATURES;   // 512 = BATCH*SEQ

    dim3 grid(OUT_FEATURES / BN, M / BM);      // 96 x 4
    dim3 block(256);
    hipLaunchKernelGGL(gq_linear_wmma_f16, grid, block, 0, stream,
                       x, w_packed, w_scale, w_bias, out, M);
}
